// Transformer_layer_78237124264435
// MI455X (gfx1250) — matrix operand order repaired and checked by reading, not compiled, not run on hardware
//
#include <hip/hip_runtime.h>
#include <hip/hip_bf16.h>
#include <math.h>

typedef unsigned short u16b;
typedef __attribute__((ext_vector_type(16))) __bf16 v16bf;
typedef __attribute__((ext_vector_type(8)))  float  v8f;
typedef __attribute__((ext_vector_type(4)))  unsigned int u32x4;
typedef __attribute__((ext_vector_type(8)))  unsigned int u32x8;

#define Bc 4
#define Lc 2048
#define Dc 1024
#define Hc 16
#define HDc 64
#define INNERc 1024
#define FFc 4096

static __device__ __forceinline__ u16b f32_to_bf16(float f) {
  unsigned int u = __float_as_uint(f);
  unsigned int r = (u + 0x7FFFu + ((u >> 16) & 1u)) >> 16;  // RNE
  return (u16b)r;
}

union FragBf {
  v16bf v;
  uint4 q[2];
  u16b  s[16];
};

static __device__ __forceinline__ v8f wmma_bf16(v16bf a, v16bf b, v8f c) {
  return __builtin_amdgcn_wmma_f32_16x16x32_bf16(false, a, false, b, (short)0, c,
                                                 false, false);
}

static __device__ __forceinline__ v8f v8f_zero() {
  v8f z;
#pragma unroll
  for (int i = 0; i < 8; ++i) z[i] = 0.0f;
  return z;
}

static __device__ __forceinline__ void lds_fence() {
  __builtin_amdgcn_wave_barrier();
  asm volatile("s_wait_dscnt 0" ::: "memory");
  __builtin_amdgcn_wave_barrier();
}

// ---------------------------------------------------------------------------
// Tensor Data Mover: DMA a 2D tile of bf16 (tile_w elems x tile_h rows, row
// stride row_stride elems) from global memory into LDS at lds_addr.
// Builds the D# (group0 + group1) per CDNA5 ISA ch.8 and issues
// TENSOR_LOAD_TO_LDS; completion tracked with TENSORcnt.
// ---------------------------------------------------------------------------
static __device__ __forceinline__ void tdm_load_2d(unsigned int lds_addr,
                                                   const void* gptr,
                                                   unsigned int tile_w,
                                                   unsigned int tile_h,
                                                   unsigned int row_stride) {
  unsigned long long ga = (unsigned long long)(uintptr_t)gptr;
  u32x4 g0;
  g0[0] = 1u;                    // count=1, is_restore=0, gather off
  g0[1] = lds_addr;              // LDS byte address
  g0[2] = (unsigned int)ga;      // global_addr[31:0]
  g0[3] = ((unsigned int)(ga >> 32) & 0x01FFFFFFu) | 0x80000000u;  // type=2
  const unsigned int td0 = row_stride;   // tensor_dim0 (elems)
  const unsigned int td1 = 1u << 20;     // tensor_dim1 (large; OOB off)
  u32x8 g1;
  g1[0] = 0x00010000u;                          // wg_mask=0, data_size=2B
  g1[1] = (td0 & 0xFFFFu) << 16;                // atomic_bar=0 | td0[15:0]
  g1[2] = (td0 >> 16) | ((td1 & 0xFFFFu) << 16);// td0[31:16] | td1[15:0]
  g1[3] = (td1 >> 16) | (tile_w << 16);         // td1[31:16] | tile_dim0
  g1[4] = tile_h;                               // tile_dim1 | tile_dim2=0
  g1[5] = row_stride;                           // dim0_stride[31:0]
  g1[6] = 0u;                                   // dim0_stride[47:32] | d1s lo
  g1[7] = 0u;
  asm volatile("tensor_load_to_lds %0, %1" ::"s"(g0), "s"(g1) : "memory");
}

static __device__ __forceinline__ void tdm_wait() {
  __builtin_amdgcn_s_wait_tensorcnt(0);
}

// ---------------------------------------------------------------------------
// f32 -> bf16 elementwise convert (4 elems/thread)
// ---------------------------------------------------------------------------
__global__ __launch_bounds__(256) void cvt_f32_bf16_kernel(
    const float* __restrict__ in, u16b* __restrict__ out, int n) {
  int i = (blockIdx.x * 256 + threadIdx.x) * 4;
  if (i + 3 < n) {
    float4 v = *(const float4*)(in + i);
    out[i + 0] = f32_to_bf16(v.x);
    out[i + 1] = f32_to_bf16(v.y);
    out[i + 2] = f32_to_bf16(v.z);
    out[i + 3] = f32_to_bf16(v.w);
  }
}

// ---------------------------------------------------------------------------
// Weight transpose+convert: wt[n*K+k] = bf16(w[k*N+n]); grid (K/32, N/32)
// ---------------------------------------------------------------------------
__global__ __launch_bounds__(256) void transpose_cvt_kernel(
    const float* __restrict__ w, u16b* __restrict__ wt, int K, int N) {
  __shared__ float tile[32][33];
  int k0 = blockIdx.x * 32, n0 = blockIdx.y * 32;
  int tx = threadIdx.x, ty = threadIdx.y;  // 32 x 8
#pragma unroll
  for (int i = 0; i < 32; i += 8)
    tile[ty + i][tx] = w[(size_t)(k0 + ty + i) * N + (n0 + tx)];
  __syncthreads();
#pragma unroll
  for (int i = 0; i < 32; i += 8)
    wt[(size_t)(n0 + ty + i) * K + (k0 + tx)] = f32_to_bf16(tile[tx][ty + i]);
}

// ---------------------------------------------------------------------------
// LayerNorm over D=1024, one row per block (256 threads), writes bf16.
// ---------------------------------------------------------------------------
__global__ __launch_bounds__(256) void ln_bf16_kernel(
    const float* __restrict__ x, const float* __restrict__ g,
    const float* __restrict__ bta, u16b* __restrict__ out) {
  __shared__ float red[8];
  int row = blockIdx.x;
  const float* xr = x + (size_t)row * Dc;
  float v[4];
  float s = 0.0f;
#pragma unroll
  for (int i = 0; i < 4; ++i) {
    v[i] = xr[threadIdx.x + i * 256];
    s += v[i];
  }
#pragma unroll
  for (int off = 16; off >= 1; off >>= 1) s += __shfl_xor(s, off, 32);
  if ((threadIdx.x & 31) == 0) red[threadIdx.x >> 5] = s;
  __syncthreads();
  float mu = 0.0f;
#pragma unroll
  for (int i = 0; i < 8; ++i) mu += red[i];
  mu *= (1.0f / Dc);
  __syncthreads();
  float var = 0.0f;
#pragma unroll
  for (int i = 0; i < 4; ++i) {
    float d = v[i] - mu;
    var += d * d;
  }
#pragma unroll
  for (int off = 16; off >= 1; off >>= 1) var += __shfl_xor(var, off, 32);
  if ((threadIdx.x & 31) == 0) red[threadIdx.x >> 5] = var;
  __syncthreads();
  float vv = 0.0f;
#pragma unroll
  for (int i = 0; i < 8; ++i) vv += red[i];
  vv *= (1.0f / Dc);
  float rinv = rsqrtf(vv + 1e-5f);
#pragma unroll
  for (int i = 0; i < 4; ++i) {
    int c = threadIdx.x + i * 256;
    out[(size_t)row * Dc + c] = f32_to_bf16((v[i] - mu) * rinv * g[c] + bta[c]);
  }
}

// ---------------------------------------------------------------------------
// bf16 WMMA GEMM: C[M x N] = A[M x K] @ Bt[N x K]^T + bias, with epilogues.
// Block = 256 threads (8 waves); wave computes 32 rows x 64 cols.
// B tile (64 cols x 32 K, 4 KB) is TDM-staged into LDS, double-buffered;
// wave 0 drives the DMA, everyone reads fragments from LDS.
// grid = (N/64, M/256).
// EPI: 0 = bf16 out, 1 = KV split (K natural bf16, V transposed [b,h,hd,L]),
//      2 = f32 out + residual, 3 = exact-GELU bf16 out.
// ---------------------------------------------------------------------------
template <int EPI>
__global__ __launch_bounds__(256) void gemm_bf16_kernel(
    const u16b* __restrict__ A, const u16b* __restrict__ Bt,
    const float* __restrict__ bias, const float* __restrict__ res,
    u16b* __restrict__ outBf, float* __restrict__ outF,
    u16b* __restrict__ outV, int M, int N, int K) {
  __shared__ __align__(16) u16b BtLds[2][64 * 32];
  int lane = threadIdx.x & 31;
  int wave = threadIdx.x >> 5;
  int lo = lane & 15, hi = lane >> 4;
  int rowBase = blockIdx.y * 256 + wave * 32;
  int colBase = blockIdx.x * 64;

  unsigned int ldsA0 = (unsigned int)(uintptr_t)(void*)&BtLds[0][0];
  unsigned int ldsA1 = (unsigned int)(uintptr_t)(void*)&BtLds[1][0];

  v8f acc[2][4];
#pragma unroll
  for (int mi = 0; mi < 2; ++mi)
#pragma unroll
    for (int ni = 0; ni < 4; ++ni) acc[mi][ni] = v8f_zero();

  const u16b* gB = Bt + (size_t)colBase * K;
  if (wave == 0) tdm_load_2d(ldsA0, gB, 32, 64, (unsigned int)K);

  int nSteps = K / 32;
  for (int idx = 0; idx < nSteps; ++idx) {
    int k0 = idx * 32;
    int buf = idx & 1;
    if (wave == 0) tdm_wait();
    __syncthreads();  // tile `buf` visible; previous tile reads all retired
    if (wave == 0 && idx + 1 < nSteps)
      tdm_load_2d(buf ? ldsA0 : ldsA1, gB + (k0 + 32), 32, 64,
                  (unsigned int)K);

    FragBf a[2];
#pragma unroll
    for (int mi = 0; mi < 2; ++mi) {
      const u16b* ap = A + (size_t)(rowBase + mi * 16 + lo) * K + k0 + hi * 8;
      a[mi].q[0] = *(const uint4*)ap;
      a[mi].q[1] = *(const uint4*)(ap + 16);
    }
    const u16b* btile = &BtLds[buf][0];
#pragma unroll
    for (int ni = 0; ni < 4; ++ni) {
      const u16b* bp = btile + (ni * 16 + lo) * 32 + hi * 8;
      FragBf bfr;
      bfr.q[0] = *(const uint4*)bp;
      bfr.q[1] = *(const uint4*)(bp + 16);
      acc[0][ni] = wmma_bf16(a[0].v, bfr.v, acc[0][ni]);
      acc[1][ni] = wmma_bf16(a[1].v, bfr.v, acc[1][ni]);
    }
  }

#pragma unroll
  for (int mi = 0; mi < 2; ++mi)
#pragma unroll
    for (int ni = 0; ni < 4; ++ni)
#pragma unroll
      for (int r = 0; r < 8; ++r) {
        int row = rowBase + mi * 16 + r + 8 * hi;
        int col = colBase + ni * 16 + lo;
        float v = acc[mi][ni][r] + bias[col];
        if constexpr (EPI == 0) {
          outBf[(size_t)row * N + col] = f32_to_bf16(v);
        } else if constexpr (EPI == 1) {
          if (col < INNERc) {
            outBf[(size_t)row * INNERc + col] = f32_to_bf16(v);
          } else {
            int c2 = col - INNERc;
            int hh = c2 >> 6, hd = c2 & 63;
            int bb = row >> 11, l = row & (Lc - 1);
            outV[((((size_t)bb * Hc + hh) * HDc + hd) << 11) + l] =
                f32_to_bf16(v);
          }
        } else if constexpr (EPI == 2) {
          outF[(size_t)row * N + col] = v + res[(size_t)row * N + col];
        } else {
          float gl = 0.5f * v * (1.0f + erff(v * 0.70710678118654752f));
          outBf[(size_t)row * N + col] = f32_to_bf16(gl);
        }
      }
}

// ---------------------------------------------------------------------------
// Flash attention. Q,K: bf16 [B,L,INNER]; Vt: bf16 [B,H,HD,L]; out bf16
// [B,L,INNER]. grid = (L/64, B*H), block = 128 (4 waves); each wave owns 16
// query rows. K and V 64x64 tiles are TDM-staged into LDS, DOUBLE-BUFFERED so
// the Tensor Data Mover DMA for tile j+1 overlaps the 16 WMMAs of tile j
// (one workgroup barrier per kv step). P is transposed through per-wave LDS.
// ---------------------------------------------------------------------------
__global__ __launch_bounds__(128) void flash_attn_kernel(
    const u16b* __restrict__ Qb, const u16b* __restrict__ Kb,
    const u16b* __restrict__ Vt, u16b* __restrict__ Ob) {
  __shared__ __align__(16) u16b KL[2][64 * 64];
  __shared__ __align__(16) u16b VL[2][64 * 64];
  __shared__ __align__(16) u16b Plds[4][16 * 64];
  const float scale = 0.125f;  // HD^-0.5
  int lane = threadIdx.x & 31;
  int w = threadIdx.x >> 5;
  int lo = lane & 15, hi = lane >> 4;
  int bh = blockIdx.y;
  int bb = bh / Hc, h = bh % Hc;
  int qr0 = blockIdx.x * 64 + w * 16;

  unsigned int klAddr[2], vlAddr[2];
  klAddr[0] = (unsigned int)(uintptr_t)(void*)&KL[0][0];
  klAddr[1] = (unsigned int)(uintptr_t)(void*)&KL[1][0];
  vlAddr[0] = (unsigned int)(uintptr_t)(void*)&VL[0][0];
  vlAddr[1] = (unsigned int)(uintptr_t)(void*)&VL[1][0];
  const u16b* Kg = Kb + (size_t)bb * Lc * INNERc + h * HDc;  // row kv
  const u16b* Vg = Vt + (size_t)bh * HDc * Lc;               // row hd

  // Q fragments (A-matrix layout), 16 rows x 64 cols -> 2 k-steps of 32
  FragBf qf[2];
  const u16b* Qbase = Qb + ((size_t)bb * Lc + qr0 + lo) * INNERc + h * HDc;
#pragma unroll
  for (int ks = 0; ks < 2; ++ks) {
    int c = ks * 32 + hi * 8;
    qf[ks].q[0] = *(const uint4*)(Qbase + c);
    qf[ks].q[1] = *(const uint4*)(Qbase + c + 16);
  }

  v8f o[4];
#pragma unroll
  for (int ni = 0; ni < 4; ++ni) o[ni] = v8f_zero();
  float m_r[8], l_r[8];
#pragma unroll
  for (int r = 0; r < 8; ++r) {
    m_r[r] = -3.0e38f;
    l_r[r] = 0.0f;
  }

  const int nTiles = Lc / 64;
  if (w == 0) {
    tdm_load_2d(klAddr[0], Kg, 64, 64, INNERc);
    tdm_load_2d(vlAddr[0], Vg, 64, 64, Lc);
  }

  for (int j = 0; j < nTiles; ++j) {
    int buf = j & 1;
    if (w == 0) tdm_wait();
    __syncthreads();  // tiles `buf` visible; reads of buf^1 all retired
    if (w == 0 && j + 1 < nTiles) {
      int kv1 = (j + 1) * 64;
      tdm_load_2d(klAddr[buf ^ 1], Kg + (size_t)kv1 * INNERc, 64, 64, INNERc);
      tdm_load_2d(vlAddr[buf ^ 1], Vg + kv1, 64, 64, Lc);
    }

    // ---- S = Q @ K^T : 4 kv sub-tiles of 16, 2 k-steps each -> 8 wmma
    v8f s[4];
#pragma unroll
    for (int ni = 0; ni < 4; ++ni) {
      s[ni] = v8f_zero();
#pragma unroll
      for (int ks = 0; ks < 2; ++ks) {
        const u16b* kp = &KL[buf][(ni * 16 + lo) * 64 + ks * 32 + hi * 8];
        FragBf kfr;
        kfr.q[0] = *(const uint4*)kp;
        kfr.q[1] = *(const uint4*)(kp + 16);
        s[ni] = wmma_bf16(qf[ks].v, kfr.v, s[ni]);
      }
    }
    // ---- online softmax (rows live across the 16 lanes of each half-wave)
    float alpha[8];
#pragma unroll
    for (int r = 0; r < 8; ++r) {
      float mx = fmaxf(fmaxf(s[0][r], s[1][r]), fmaxf(s[2][r], s[3][r]));
#pragma unroll
      for (int off = 8; off >= 1; off >>= 1)
        mx = fmaxf(mx, __shfl_xor(mx, off, 32));
      float mnew = fmaxf(m_r[r], mx);
      alpha[r] = __expf(scale * (m_r[r] - mnew));
      m_r[r] = mnew;
    }
    float rowsum[8];
#pragma unroll
    for (int r = 0; r < 8; ++r) rowsum[r] = 0.0f;
#pragma unroll
    for (int ni = 0; ni < 4; ++ni)
#pragma unroll
      for (int r = 0; r < 8; ++r) {
        float p = __expf(scale * (s[ni][r] - m_r[r]));
        rowsum[r] += p;
        Plds[w][(r + 8 * hi) * 64 + ni * 16 + lo] = f32_to_bf16(p);
      }
#pragma unroll
    for (int r = 0; r < 8; ++r) {
#pragma unroll
      for (int off = 8; off >= 1; off >>= 1)
        rowsum[r] += __shfl_xor(rowsum[r], off, 32);
      l_r[r] = l_r[r] * alpha[r] + rowsum[r];
    }
#pragma unroll
    for (int ni = 0; ni < 4; ++ni)
#pragma unroll
      for (int r = 0; r < 8; ++r) o[ni][r] *= alpha[r];

    lds_fence();  // P writes visible to all lanes of this wave

    // ---- O += P @ V : P as A-frag from LDS, V rows (hd) contiguous along kv
#pragma unroll
    for (int ks = 0; ks < 2; ++ks) {
      FragBf pf;
      const u16b* pptr = &Plds[w][lo * 64 + ks * 32 + hi * 8];
      pf.q[0] = *(const uint4*)pptr;
      pf.q[1] = *(const uint4*)(pptr + 16);
#pragma unroll
      for (int ni = 0; ni < 4; ++ni) {
        const u16b* vp = &VL[buf][(ni * 16 + lo) * 64 + ks * 32 + hi * 8];
        FragBf vf;
        vf.q[0] = *(const uint4*)vp;
        vf.q[1] = *(const uint4*)(vp + 16);
        o[ni] = wmma_bf16(pf.v, vf.v, o[ni]);
      }
    }
  }

  // ---- normalize + store bf16 [b, l, h*64+hd]
  float inv[8];
#pragma unroll
  for (int r = 0; r < 8; ++r) inv[r] = 1.0f / l_r[r];
#pragma unroll
  for (int ni = 0; ni < 4; ++ni)
#pragma unroll
    for (int r = 0; r < 8; ++r) {
      int row = qr0 + r + 8 * hi;
      int col = h * HDc + ni * 16 + lo;
      Ob[((size_t)bb * Lc + row) * INNERc + col] =
          f32_to_bf16(o[ni][r] * inv[r]);
    }
}

// ---------------------------------------------------------------------------
extern "C" void kernel_launch(void* const* d_in, const int* in_sizes, int n_in,
                              void* d_out, int out_size, void* d_ws,
                              size_t ws_size, hipStream_t stream) {
  const float* x_q = (const float*)d_in[0];
  const float* x_kv = (const float*)d_in[1];
  const float* ln1_g = (const float*)d_in[2];
  const float* ln1_b = (const float*)d_in[3];
  const float* wq = (const float*)d_in[4];
  const float* bq = (const float*)d_in[5];
  const float* wkv = (const float*)d_in[6];
  const float* bkv = (const float*)d_in[7];
  const float* wo = (const float*)d_in[8];
  const float* bo = (const float*)d_in[9];
  const float* ln2_g = (const float*)d_in[10];
  const float* ln2_b = (const float*)d_in[11];
  const float* w1 = (const float*)d_in[12];
  const float* b1 = (const float*)d_in[13];
  const float* w2 = (const float*)d_in[14];
  const float* b2 = (const float*)d_in[15];
  float* out = (float*)d_out;

  const int M = Bc * Lc;  // 8192
  char* cur = (char*)d_ws;
  auto alloc = [&](size_t bytes) -> char* {
    char* p = cur;
    cur += (bytes + 255) & ~(size_t)255;
    return p;
  };
  u16b* xn1 = (u16b*)alloc((size_t)M * Dc * 2);
  u16b* xkvb = (u16b*)alloc((size_t)M * Dc * 2);
  u16b* wq_t = (u16b*)alloc((size_t)INNERc * Dc * 2);
  u16b* wkv_t = (u16b*)alloc((size_t)2 * INNERc * Dc * 2);
  u16b* wo_t = (u16b*)alloc((size_t)Dc * INNERc * 2);
  u16b* w1_t = (u16b*)alloc((size_t)FFc * Dc * 2);
  u16b* w2_t = (u16b*)alloc((size_t)Dc * FFc * 2);
  u16b* qb = (u16b*)alloc((size_t)M * INNERc * 2);
  u16b* kb = (u16b*)alloc((size_t)M * INNERc * 2);
  u16b* vtb = (u16b*)alloc((size_t)M * INNERc * 2);
  u16b* aob = (u16b*)alloc((size_t)M * INNERc * 2);
  float* x1 = (float*)alloc((size_t)M * Dc * 4);
  u16b* xn2 = (u16b*)alloc((size_t)M * Dc * 2);
  u16b* hb = (u16b*)alloc((size_t)M * FFc * 2);

  dim3 tb(32, 8);
  transpose_cvt_kernel<<<dim3(Dc / 32, INNERc / 32), tb, 0, stream>>>(
      wq, wq_t, Dc, INNERc);
  transpose_cvt_kernel<<<dim3(Dc / 32, (2 * INNERc) / 32), tb, 0, stream>>>(
      wkv, wkv_t, Dc, 2 * INNERc);
  transpose_cvt_kernel<<<dim3(INNERc / 32, Dc / 32), tb, 0, stream>>>(
      wo, wo_t, INNERc, Dc);
  transpose_cvt_kernel<<<dim3(Dc / 32, FFc / 32), tb, 0, stream>>>(
      w1, w1_t, Dc, FFc);
  transpose_cvt_kernel<<<dim3(FFc / 32, Dc / 32), tb, 0, stream>>>(
      w2, w2_t, FFc, Dc);

  ln_bf16_kernel<<<M, 256, 0, stream>>>(x_q, ln1_g, ln1_b, xn1);
  cvt_f32_bf16_kernel<<<(M * Dc) / 1024, 256, 0, stream>>>(x_kv, xkvb, M * Dc);

  // Q = xn1 @ wq + bq  (bf16 out)
  gemm_bf16_kernel<0><<<dim3(INNERc / 64, M / 256), 256, 0, stream>>>(
      xn1, wq_t, bq, nullptr, qb, nullptr, nullptr, M, INNERc, Dc);
  // KV = xkv @ wkv + bkv  (K natural, V transposed)
  gemm_bf16_kernel<1><<<dim3((2 * INNERc) / 64, M / 256), 256, 0, stream>>>(
      xkvb, wkv_t, bkv, nullptr, kb, nullptr, vtb, M, 2 * INNERc, Dc);

  // attention
  flash_attn_kernel<<<dim3(Lc / 64, Bc * Hc), 128, 0, stream>>>(qb, kb, vtb,
                                                                aob);

  // x1 = attn @ wo + bo + x_q  (f32)
  gemm_bf16_kernel<2><<<dim3(Dc / 64, M / 256), 256, 0, stream>>>(
      aob, wo_t, bo, x_q, nullptr, x1, nullptr, M, Dc, INNERc);

  // LN2
  ln_bf16_kernel<<<M, 256, 0, stream>>>(x1, ln2_g, ln2_b, xn2);

  // h = gelu(xn2 @ w1 + b1)  (bf16)
  gemm_bf16_kernel<3><<<dim3(FFc / 64, M / 256), 256, 0, stream>>>(
      xn2, w1_t, b1, nullptr, hb, nullptr, nullptr, M, FFc, Dc);

  // out = h @ w2 + b2 + x1  (f32)
  gemm_bf16_kernel<2><<<dim3(Dc / 64, M / 256), 256, 0, stream>>>(
      hb, w2_t, b2, x1, nullptr, out, nullptr, M, Dc, FFc);
}
